// MC_net_10_3719441678866
// MI455X (gfx1250) — 8-bit matrix operand order repaired and checked by reading, compiled, not run on hardware
//
#include <hip/hip_runtime.h>


typedef __attribute__((ext_vector_type(8))) int      v8i;
typedef __attribute__((ext_vector_type(4))) int      v4i;
typedef __attribute__((ext_vector_type(4))) unsigned v4u;

#define BN_RSQRT 0.99999500003749968f

#if __has_builtin(__builtin_amdgcn_tensor_load_to_lds)
#define HAVE_TDM 1
#else
#define HAVE_TDM 0
#endif
#if __has_include(<hip/amd_detail/amd_gfx1250_TDM.h>)
#define TDM_SIX 1
#else
#define TDM_SIX 0
#endif

#if HAVE_TDM
__device__ __forceinline__ void tdm_load_2d(unsigned ldsOff, const void* gptr,
                                            unsigned remW, unsigned remH,
                                            unsigned tileW, unsigned tileH,
                                            unsigned long long rowStride) {
  unsigned long long ga = (unsigned long long)(size_t)gptr;
  v4u g0 = {0u, 0u, 0u, 0u};
  g0[0] = 1u;
  g0[1] = ldsOff;
  g0[2] = (unsigned)ga;
  g0[3] = (unsigned)((ga >> 32) & 0x1ffffffull)
          | (2u << 30);
  v8i g1 = {0, 0, 0, 0, 0, 0, 0, 0};
  g1[0] = 0;
  g1[1] = (int)(remW << 16);
  g1[2] = (int)((remW >> 16) | (remH << 16));
  g1[3] = (int)((remH >> 16) | (tileW << 16));
  g1[4] = (int)tileH;
  g1[5] = (int)(rowStride & 0xffffffffull);
  g1[6] = (int)((rowStride >> 32) & 0xffffull);
  g1[7] = 0;
  v4i z4 = {0, 0, 0, 0};
#if TDM_SIX
  v8i z8 = {0, 0, 0, 0, 0, 0, 0, 0};
  __builtin_amdgcn_tensor_load_to_lds(g0, g1, z4, z4, z8, 0);
#else
  __builtin_amdgcn_tensor_load_to_lds(g0, g1, z4, z4, 0);
#endif
}
#endif


__global__ void __launch_bounds__(256)
k_binarize(const float* __restrict__ x, signed char* __restrict__ y, long long n) {
  long long i = (long long)blockIdx.x * blockDim.x + threadIdx.x;
  if (i >= n) return;
  float v = x[i];
  y[i] = (v > 0.f) ? (signed char)1 : ((v < 0.f) ? (signed char)-1 : (signed char)0);
}

__global__ void __launch_bounds__(256)
k_add(const float* __restrict__ A, const float* __restrict__ Bb,
      float* __restrict__ Y, int N, int C, int H, int W, int Hb) {
  long long total = (long long)N * C * H * W;
  long long i = (long long)blockIdx.x * blockDim.x + threadIdx.x;
  if (i >= total) return;
  int w = (int)(i % W);
  long long t = i / W;
  int h = (int)(t % H); t /= H;
  int c = (int)(t % C);
  long long n = t / C;
  int hb = (Hb == H) ? h : 0;
  Y[i] = A[i] + Bb[((n * C + c) * Hb + hb) * W + w];
}

__global__ void __launch_bounds__(256)
k_copych(const float* __restrict__ X, float* __restrict__ Y,
         int N, int C, int H, int W, int Ctot, int cOff) {
  long long total = (long long)N * C * H * W;
  long long i = (long long)blockIdx.x * blockDim.x + threadIdx.x;
  if (i >= total) return;
  int w = (int)(i % W);
  long long t = i / W;
  int h = (int)(t % H); t /= H;
  int c = (int)(t % C);
  long long n = t / C;
  Y[((n * Ctot + cOff + c) * H + h) * W + w] = X[i];
}


__global__ void __launch_bounds__(256)
k_maxpool(const float* __restrict__ X, float* __restrict__ Y,
          int N, int C, int H, int W, int H2, int W2,
          int kh, int kw, int sh, int sw, int ph, int pw,
          int Ctot, int cOff) {
  long long total = (long long)N * C * H2 * W2;
  long long i = (long long)blockIdx.x * blockDim.x + threadIdx.x;
  if (i >= total) return;
  int w2 = (int)(i % W2);
  long long t = i / W2;
  int h2 = (int)(t % H2); t /= H2;
  int c = (int)(t % C);
  long long n = t / C;
  float m = -__builtin_inff();
  for (int ky = 0; ky < kh; ++ky) {
    int h = h2 * sh - ph + ky;
    if (h < 0 || h >= H) continue;
    for (int kx = 0; kx < kw; ++kx) {
      int w = w2 * sw - pw + kx;
      if (w < 0 || w >= W) continue;
      m = fmaxf(m, X[((n * C + c) * H + h) * W + w]);
    }
  }
  Y[((n * Ctot + cOff + c) * H2 + h2) * W2 + w2] = m;
}

__global__ void __launch_bounds__(256)
k_avgpool(const float* __restrict__ X, float* __restrict__ Y,
          int N, int C, int H, int W, int H2, int W2,
          int kh, int kw, int sh, int sw, int ph, int pw,
          int Ctot, int cOff) {
  long long total = (long long)N * C * H2 * W2;
  long long i = (long long)blockIdx.x * blockDim.x + threadIdx.x;
  if (i >= total) return;
  int w2 = (int)(i % W2);
  long long t = i / W2;
  int h2 = (int)(t % H2); t /= H2;
  int c = (int)(t % C);
  long long n = t / C;
  float s = 0.f;
  for (int ky = 0; ky < kh; ++ky) {
    int h = h2 * sh - ph + ky;
    if (h < 0 || h >= H) continue;
    for (int kx = 0; kx < kw; ++kx) {
      int w = w2 * sw - pw + kx;
      if (w < 0 || w >= W) continue;
      s += X[((n * C + c) * H + h) * W + w];
    }
  }
  Y[((n * Ctot + cOff + c) * H2 + h2) * W2 + w2] = s / (float)(kh * kw);
}

__global__ void __launch_bounds__(256)
k_jumppool(const float* __restrict__ X, float* __restrict__ Y,
           int N, int C, int W, int W2) {
  long long total = (long long)N * C * W2;
  long long i = (long long)blockIdx.x * blockDim.x + threadIdx.x;
  if (i >= total) return;
  int ow = (int)(i % W2);
  long long t = i / W2;
  int c = (int)(t % C);
  long long n = t / C;
  const float* base = X + ((n * C + c) * 2) * (long long)W;
  float m = -__builtin_inff();
  for (int h = 0; h < 2; ++h)
    for (int d = 0; d < 2; ++d) {
      int w = 2 * ow - 1 + d;
      float v = (w >= 0) ? base[(long long)h * W + w] : 0.f;
      m = fmaxf(m, v);
    }
  Y[i] = m;
}

__global__ void __launch_bounds__(32)
k_bconv_wmma(const signed char* __restrict__ X, const signed char* __restrict__ Wt,
             float* __restrict__ Y,
             const float* __restrict__ scale, const float* __restrict__ bias,
             int N, int Cin, int H, int W,
             int Cout, int kh, int kw, int sh, int sw, int ph, int pw,
             int H2, int W2, int CtotOut, int cOff, int doRelu) {
  __shared__ __align__(16) signed char ldsA[16 * 64];
  __shared__ __align__(16) signed char ldsB[32 * 64];

  const int lane = threadIdx.x;
  const int half = lane >> 4;
  const int l15  = lane & 15;

  const int K  = Cin * kh * kw;
  const int P  = N * H2 * W2;
  const int khkw = kh * kw;
  const int posTile  = blockIdx.x << 5;
  const int coutTile = blockIdx.y << 4;

  v8i acc0 = {0, 0, 0, 0, 0, 0, 0, 0};
  v8i acc1 = {0, 0, 0, 0, 0, 0, 0, 0};

  const int kSteps = (K + 63) >> 6;
  for (int ks = 0; ks < kSteps; ++ks) {
    const int kb = ks << 6;

#if HAVE_TDM
    tdm_load_2d((unsigned)(size_t)ldsA,
                Wt + (long long)coutTile * K + kb,
                (unsigned)(K - kb), (unsigned)(Cout - coutTile),
                64u, 16u, (unsigned long long)K);
#else
    #pragma unroll 4
    for (int j = 0; j < 32; ++j) {
      int t = lane * 32 + j;
      int m = t >> 6, k = t & 63;
      int kk = kb + k;
      int co = coutTile + m;
      signed char v = 0;
      if (kk < K && co < Cout) v = Wt[(long long)co * K + kk];
      ldsA[t] = v;
    }
#endif

    #pragma unroll 2
    for (int j = 0; j < 64; ++j) {
      int t = lane * 64 + j;
      int n = t >> 6, k = t & 63;
      int kk = kb + k;
      int pos = posTile + n;
      signed char v = 0;
      if (kk < K && pos < P) {
        int nb = pos / (H2 * W2);
        int rp = pos - nb * (H2 * W2);
        int h2 = rp / W2, w2 = rp - h2 * W2;
        int cin = kk / khkw;
        int rk = kk - cin * khkw;
        int ky = rk / kw, kx = rk - ky * kw;
        int h = h2 * sh - ph + ky;
        int w = w2 * sw - pw + kx;
        if (h >= 0 && h < H && w >= 0 && w < W)
          v = X[(((long long)nb * Cin + cin) * H + h) * W + w];
      }
      ldsB[t] = v;
    }
    if (ks + 1 < kSteps) {
      int co = coutTile + l15;
      if (co < Cout && (kb + 64) < K)
        __builtin_prefetch(&Wt[(long long)co * K + kb + 64], 0, 1);
    }
    __syncthreads();
#if HAVE_TDM && __has_builtin(__builtin_amdgcn_s_wait_tensorcnt)
    __builtin_amdgcn_s_wait_tensorcnt(0);
#endif

    v8i a, b0, b1;
    #pragma unroll
    for (int vi = 0; vi < 8; ++vi) {
      int ka = 32 * (vi >> 2) + 16 * ((vi >> 1) & 1) + 4 * (vi & 1) + 8 * half;
      a[vi] = *(const int*)&ldsA[(l15 << 6) + ka];
      int kbf = 32 * (vi >> 2) + 16 * ((vi >> 1) & 1) + 4 * (vi & 1) + 8 * half;
      b0[vi] = *(const int*)&ldsB[(l15 << 6) + kbf];
      b1[vi] = *(const int*)&ldsB[((16 + l15) << 6) + kbf];
    }
    acc0 = __builtin_amdgcn_wmma_i32_16x16x64_iu8(true, a, true, b0, acc0, false, false);
    acc1 = __builtin_amdgcn_wmma_i32_16x16x64_iu8(true, a, true, b1, acc1, false, false);
    __syncthreads();
  }

  #pragma unroll
  for (int q = 0; q < 2; ++q) {
    const int pos = posTile + q * 16 + l15;
    if (pos < P) {
      int nb = pos / (H2 * W2);
      int rp = pos - nb * (H2 * W2);
      int h2 = rp / W2, w2 = rp - h2 * W2;
      #pragma unroll
      for (int r = 0; r < 8; ++r) {
        int co = coutTile + half * 8 + r;
        if (co < Cout) {
          float v = (float)(q ? acc1[r] : acc0[r]);
          if (scale) v = v * (scale[co] * BN_RSQRT) + bias[co];
          if (doRelu) v = v > 0.f ? v : 0.f;
          Y[(((long long)nb * CtotOut + cOff + co) * H2 + h2) * W2 + w2] = v;
        }
      }
    }
  }
}


__global__ void __launch_bounds__(256)
k_linear(const float* __restrict__ X, const float* __restrict__ Wl,
         const float* __restrict__ bl, float* __restrict__ Y,
         int B, int F, int O) {
  int i = blockIdx.x * blockDim.x + threadIdx.x;
  if (i >= B * O) return;
  int b = i / O, o = i % O;
  float s = bl[o];
  for (int f = 0; f < F; ++f) s += X[(long long)b * F + f] * Wl[(long long)o * F + f];
  Y[i] = s;
}


struct Blk {
  const float *w[4], *g[4], *b[4];
  signed char* sw[4];
  int cin, m;
};

extern "C" void kernel_launch(void* const* d_in, const int* in_sizes, int n_in,
                              void* d_out, int out_size, void* d_ws, size_t ws_size,
                              hipStream_t stream) {
  (void)in_sizes; (void)n_in; (void)out_size; (void)ws_size;
  const int NB = 512;

  const float* x       = (const float*)d_in[0];
  const float* conv1_w = (const float*)d_in[1];
  const float* pre1_w  = (const float*)d_in[2];
  const float* pre2_w  = (const float*)d_in[3];
  const float* jump_w  = (const float*)d_in[4];
  const float* lin_w   = (const float*)d_in[5];
  const float* lin_b   = (const float*)d_in[6];

  Blk blks[10];
  {
    int ii = 7;
    for (int bi = 0; bi < 10; ++bi) {
      Blk& Bk = blks[bi];
      Bk.m = (bi == 9) ? 2 : 1;
      Bk.cin = (bi == 0) ? 64 : 128;
      for (int j = 0; j < 4; ++j) Bk.w[j] = (const float*)d_in[ii++];
      for (int j = 0; j < 4; ++j) {
        Bk.g[j] = (const float*)d_in[ii++];
        Bk.b[j] = (const float*)d_in[ii++];
      }
    }
  }

  char* wsb = (char*)d_ws;
  size_t off = 0;
  auto alloc = [&](size_t bytes) -> void* {
    void* p = wsb + off;
    off = (off + bytes + 255) & ~(size_t)255;
    return p;
  };
  auto allocF  = [&](long long n) { return (float*)alloc((size_t)n * 4); };
  auto allocI8 = [&](long long n) { return (signed char*)alloc((size_t)n); };

  auto bin = [&](const float* src, signed char* dst, long long n) {
    k_binarize<<<dim3((unsigned)((n + 255) / 256)), 256, 0, stream>>>(src, dst, n);
  };
  auto conv = [&](const signed char* Xs, int N, int Cin, int H, int W,
                  const signed char* Ws, int Cout, int kh, int kw,
                  int sh, int sw, int ph, int pw,
                  float* Y, int H2, int W2, int Ctot, int cOff,
                  const float* g, const float* b, int relu) {
    dim3 grid((unsigned)((N * H2 * W2 + 31) / 32), (unsigned)((Cout + 15) / 16));
    k_bconv_wmma<<<grid, 32, 0, stream>>>(Xs, Ws, Y, g, b, N, Cin, H, W,
                                          Cout, kh, kw, sh, sw, ph, pw,
                                          H2, W2, Ctot, cOff, relu);
  };
  auto mpool = [&](const float* X, int N, int C, int H, int W,
                   int kh, int kw, int sh, int sw, int ph, int pw,
                   float* Y, int H2, int W2, int Ctot, int cOff) {
    long long t = (long long)N * C * H2 * W2;
    k_maxpool<<<dim3((unsigned)((t + 255) / 256)), 256, 0, stream>>>(
        X, Y, N, C, H, W, H2, W2, kh, kw, sh, sw, ph, pw, Ctot, cOff);
  };
  auto apool = [&](const float* X, int N, int C, int H, int W,
                   int kh, int kw, int sh, int sw, int ph, int pw,
                   float* Y, int H2, int W2, int Ctot, int cOff) {
    long long t = (long long)N * C * H2 * W2;
    k_avgpool<<<dim3((unsigned)((t + 255) / 256)), 256, 0, stream>>>(
        X, Y, N, C, H, W, H2, W2, kh, kw, sh, sw, ph, pw, Ctot, cOff);
  };
  auto add = [&](const float* A, const float* Bp, float* Y,
                 int N, int C, int H, int W, int Hb) {
    long long t = (long long)N * C * H * W;
    k_add<<<dim3((unsigned)((t + 255) / 256)), 256, 0, stream>>>(A, Bp, Y, N, C, H, W, Hb);
  };

  signed char* sc_conv1 = allocI8(64 * 1 * 3 * 7);   bin(conv1_w, sc_conv1, 64 * 21);
  signed char* sc_pre1  = allocI8(32 * 64 * 3 * 1);  bin(pre1_w,  sc_pre1,  32 * 64 * 3);
  signed char* sc_pre2  = allocI8(32 * 64 * 1 * 3);  bin(pre2_w,  sc_pre2,  32 * 64 * 3);
  signed char* sc_jump  = allocI8(128 * 64);         bin(jump_w,  sc_jump,  128 * 64);
  for (int bi = 0; bi < 10; ++bi) {
    Blk& Bk = blks[bi];
    long long s1 = 32LL * Bk.cin;
    long long s2 = 48LL * Bk.m * 32 * 3;
    long long s3 = 48LL * Bk.m * 32 * 3;
    long long s4 = 32LL * Bk.m * 32;
    Bk.sw[0] = allocI8(s1); bin(Bk.w[0], Bk.sw[0], s1);
    Bk.sw[1] = allocI8(s2); bin(Bk.w[1], Bk.sw[1], s2);
    Bk.sw[2] = allocI8(s3); bin(Bk.w[2], Bk.sw[2], s3);
    Bk.sw[3] = allocI8(s4); bin(Bk.w[3], Bk.sw[3], s4);
  }

  auto m_block = [&](const float* X, int N, int Cin, int H, int W, Blk& Bk,
                     float* OUT, int Ctot, int cBase) {
    int m = Bk.m;
    signed char* xs_ = allocI8((long long)N * Cin * H * W);
    bin(X, xs_, (long long)N * Cin * H * W);
    float* xb = allocF((long long)N * 32 * H * W);
    conv(xs_, N, Cin, H, W, Bk.sw[0], 32, 1, 1, 1, 1, 0, 0,
         xb, H, W, 32, 0, Bk.g[0], Bk.b[0], 1);
    signed char* xbs = allocI8((long long)N * 32 * H * W);
    bin(xb, xbs, (long long)N * 32 * H * W);
    conv(xbs, N, 32, H, W, Bk.sw[1], 48 * m, 3, 1, 1, 1, 1, 0,
         OUT, H, W, Ctot, cBase + 0, Bk.g[1], Bk.b[1], 1);
    conv(xbs, N, 32, H, W, Bk.sw[2], 48 * m, 1, 3, 1, 1, 0, 1,
         OUT, H, W, Ctot, cBase + 48 * m, Bk.g[2], Bk.b[2], 1);
    conv(xbs, N, 32, H, W, Bk.sw[3], 32 * m, 1, 1, 1, 1, 0, 0,
         OUT, H, W, Ctot, cBase + 96 * m, Bk.g[3], Bk.b[3], 1);
  };
  auto m_block_p = [&](const float* X, int N, int Cin, int H, int W, Blk& Bk,
                       float* OUT, int Ctot, int cBase) {
    int m = Bk.m, W2 = W / 2;
    signed char* xs_ = allocI8((long long)N * Cin * H * W);
    bin(X, xs_, (long long)N * Cin * H * W);
    float* xb = allocF((long long)N * 32 * H * W);
    conv(xs_, N, Cin, H, W, Bk.sw[0], 32, 1, 1, 1, 1, 0, 0,
         xb, H, W, 32, 0, Bk.g[0], Bk.b[0], 1);
    signed char* xbs = allocI8((long long)N * 32 * H * W);
    bin(xb, xbs, (long long)N * 32 * H * W);
    float* t2 = allocF((long long)N * 48 * m * H * W);
    conv(xbs, N, 32, H, W, Bk.sw[1], 48 * m, 3, 1, 1, 1, 1, 0,
         t2, H, W, 48 * m, 0, Bk.g[1], Bk.b[1], 1);
    mpool(t2, N, 48 * m, H, W, 1, 3, 1, 2, 0, 1, OUT, H, W2, Ctot, cBase);
    conv(xbs, N, 32, H, W, Bk.sw[2], 48 * m, 1, 3, 1, 2, 0, 1,
         OUT, H, W2, Ctot, cBase + 48 * m, Bk.g[2], Bk.b[2], 1);
    conv(xbs, N, 32, H, W, Bk.sw[3], 32 * m, 1, 1, 1, 2, 0, 0,
         OUT, H, W2, Ctot, cBase + 96 * m, Bk.g[3], Bk.b[3], 1);
  };

  signed char* xs = allocI8(512LL * 2 * 1024);
  bin(x, xs, 512LL * 2 * 1024);
  float* c1 = allocF(512LL * 64 * 2 * 512);
  conv(xs, NB, 1, 2, 1024, sc_conv1, 64, 3, 7, 1, 2, 1, 3,
       c1, 2, 512, 64, 0, nullptr, nullptr, 1);
  float* p1 = allocF(512LL * 64 * 2 * 256);
  mpool(c1, NB, 64, 2, 512, 1, 3, 1, 2, 0, 1, p1, 2, 256, 64, 0);

  signed char* p1s = allocI8(512LL * 64 * 2 * 256);
  bin(p1, p1s, 512LL * 64 * 2 * 256);
  float* t1 = allocF(512LL * 32 * 2 * 256);
  conv(p1s, NB, 64, 2, 256, sc_pre1, 32, 3, 1, 1, 1, 1, 0,
       t1, 2, 256, 32, 0, nullptr, nullptr, 1);
  float* cat1 = allocF(512LL * 64 * 2 * 128);
  apool(t1, NB, 32, 2, 256, 1, 3, 1, 2, 0, 1, cat1, 2, 128, 64, 0);
  conv(p1s, NB, 64, 2, 256, sc_pre2, 32, 1, 3, 1, 2, 0, 1,
       cat1, 2, 128, 64, 32, nullptr, nullptr, 1);

  signed char* cat1s = allocI8(512LL * 64 * 2 * 128);
  bin(cat1, cat1s, 512LL * 64 * 2 * 128);
  float* tj = allocF(512LL * 128 * 2 * 64);
  conv(cat1s, NB, 64, 2, 128, sc_jump, 128, 1, 1, 1, 2, 0, 0,
       tj, 2, 64, 128, 0, nullptr, nullptr, 1);
  float* o1b = allocF(512LL * 128 * 2 * 32);
  mpool(tj, NB, 128, 2, 64, 1, 3, 1, 2, 0, 1, o1b, 2, 32, 128, 0);
  float* mp1 = allocF(512LL * 64 * 2 * 64);
  mpool(cat1, NB, 64, 2, 128, 1, 3, 1, 2, 0, 1, mp1, 2, 64, 64, 0);
  float* blk1o = allocF(512LL * 128 * 2 * 32);
  m_block_p(mp1, NB, 64, 2, 64, blks[0], blk1o, 128, 0);
  float* cur = allocF(512LL * 128 * 2 * 32);
  add(o1b, blk1o, cur, NB, 128, 2, 32, 2);

  int Wc = 32;
  for (int s = 0; s < 4; ++s) {
    float* tb = allocF(512LL * 128 * 2 * Wc);
    m_block(cur, NB, 128, 2, Wc, blks[1 + 2 * s], tb, 128, 0);
    float* r1 = allocF(512LL * 128 * 2 * Wc);
    add(tb, cur, r1, NB, 128, 2, Wc, 2);
    cur = r1;

    int W2 = Wc / 2;
    float* tp = allocF(512LL * 128 * 2 * W2);
    m_block_p(cur, NB, 128, 2, Wc, blks[2 + 2 * s], tp, 128, 0);
    float* jp = allocF(512LL * 128 * 1 * W2);
    {
      long long t = 512LL * 128 * W2;
      k_jumppool<<<dim3((unsigned)((t + 255) / 256)), 256, 0, stream>>>(
          cur, jp, NB, 128, Wc, W2);
    }
    float* r2 = allocF(512LL * 128 * 2 * W2);
    add(tp, jp, r2, NB, 128, 2, W2, 1);
    cur = r2;
    Wc = W2;
  }

  float* cat2 = allocF(512LL * 384 * 2 * 2);
  {
    long long t = 512LL * 128 * 2 * 2;
    k_copych<<<dim3((unsigned)((t + 255) / 256)), 256, 0, stream>>>(
        cur, cat2, NB, 128, 2, 2, 384, 0);
  }
  m_block(cur, NB, 128, 2, 2, blks[9], cat2, 384, 128);

  float* gap = allocF(512LL * 384);
  apool(cat2, NB, 384, 2, 2, 2, 2, 2, 2, 0, 0, gap, 1, 1, 384, 0);
  k_linear<<<dim3((unsigned)((512 * 24 + 255) / 256)), 256, 0, stream>>>(
      gap, lin_w, lin_b, (float*)d_out, 512, 384, 24);
}
